// TemporalGATFraudDetector_19567871001358
// MI455X (gfx1250) — matrix operand order repaired and checked by reading, not compiled, not run on hardware
//
#include <hip/hip_runtime.h>
#include <hip/hip_bf16.h>
#include <math.h>

typedef __attribute__((ext_vector_type(16))) __bf16 v16bf;
typedef __attribute__((ext_vector_type(8)))  __bf16 v8bf;
typedef __attribute__((ext_vector_type(8)))  float  v8f;

#define K_N 50000
#define K_E 800000
#define K_IN 128

// ---------------- helpers ----------------
__device__ __forceinline__ unsigned orderf(float f) {
    unsigned u = __float_as_uint(f);
    return (u & 0x80000000u) ? ~u : (u | 0x80000000u);
}
__device__ __forceinline__ float unorderf(unsigned u) {
    unsigned b = (u & 0x80000000u) ? (u & 0x7FFFFFFFu) : ~u;
    return __uint_as_float(b);
}
__device__ __forceinline__ unsigned short f2bf(float f) {
    unsigned u = __float_as_uint(f);
    u += 0x7FFFu + ((u >> 16) & 1u);   // round-to-nearest-even
    return (unsigned short)(u >> 16);
}

// ---------------- small utility kernels ----------------
__global__ void zero_f(float* __restrict__ p, int n) {
    for (int i = blockIdx.x * blockDim.x + threadIdx.x; i < n; i += gridDim.x * blockDim.x)
        p[i] = 0.0f;
}

__global__ void init_red(unsigned* red) { red[0] = 0xFFFFFFFFu; red[1] = 0u; }

__global__ void ts_minmax(const float* __restrict__ ea, unsigned* __restrict__ red, int ne) {
    __shared__ unsigned smin[256], smax[256];
    int t = threadIdx.x;
    int i = blockIdx.x * 256 + t;
    unsigned lo = 0xFFFFFFFFu, hi = 0u;
    if (i < ne) { unsigned o = orderf(ea[i * 5 + 1]); lo = o; hi = o; }
    smin[t] = lo; smax[t] = hi;
    __syncthreads();
    for (int s = 128; s > 0; s >>= 1) {
        if (t < s) {
            smin[t] = min(smin[t], smin[t + s]);
            smax[t] = max(smax[t], smax[t + s]);
        }
        __syncthreads();
    }
    if (t == 0) { atomicMin(&red[0], smin[0]); atomicMax(&red[1], smax[0]); }
}

__global__ void conv_x_bf16(const float* __restrict__ in, unsigned short* __restrict__ out, int n) {
    for (int i = blockIdx.x * blockDim.x + threadIdx.x; i < n; i += gridDim.x * blockDim.x)
        out[i] = f2bf(in[i]);
}

// W (128x128 row-major, 3 mats) -> bf16 column-major [mat][col][k]
__global__ void conv_w_bf16(const float* __restrict__ wq, const float* __restrict__ wk,
                            const float* __restrict__ wv, unsigned short* __restrict__ wbf) {
    int i = blockIdx.x * blockDim.x + threadIdx.x;
    if (i >= 3 * 128 * 128) return;
    int mat = i >> 14, rem = i & 16383;
    int col = rem >> 7, kk = rem & 127;
    const float* W = (mat == 0) ? wq : (mat == 1) ? wk : wv;
    wbf[i] = f2bf(W[kk * 128 + col]);
}

__global__ void elu_inplace(float* __restrict__ p, int n) {
    for (int i = blockIdx.x * blockDim.x + threadIdx.x; i < n; i += gridDim.x * blockDim.x) {
        float x = p[i];
        p[i] = x > 0.0f ? x : expm1f(x);
    }
}

// ---------------- WMMA GEMM: [N x 128] @ [128 x 128] for q/k/v ----------------
__global__ void gemm_qkv_wmma(const unsigned short* __restrict__ xbf,
                              const unsigned short* __restrict__ wbf,
                              float* __restrict__ q, float* __restrict__ k,
                              float* __restrict__ v) {
    const int m0   = blockIdx.x * 16;
    const int mat  = blockIdx.y;          // 0=q 1=k 2=v
    const int wave = threadIdx.x >> 5;    // 8 waves -> 8 column tiles
    const int lane = threadIdx.x & 31;
    const int n0   = wave * 16;
    const unsigned short* W = wbf + mat * 128 * 128;
    float* outp = (mat == 0) ? q : (mat == 1) ? k : v;

    const int mr   = lane & 15;
    const int ksel = lane >> 4;
    const unsigned short* arow = xbf + (size_t)(m0 + mr) * 128;
    const unsigned short* bcol = W + (size_t)(n0 + mr) * 128;

    v8f acc = {};
#pragma unroll
    for (int kc = 0; kc < 4; ++kc) {
        const int k0 = kc * 32;
        // A fragment (16-bit A 16x32 layout)
        v8bf alo = *(const v8bf*)(arow + k0 + ksel * 8);        // K: k0+{0..7} / {8..15}
        v8bf ahi = *(const v8bf*)(arow + k0 + 16 + ksel * 8);   // K: k0+{16..23} / {24..31}
        v16bf a;
#pragma unroll
        for (int i = 0; i < 8; ++i) { a[i] = alo[i]; a[i + 8] = ahi[i]; }
        // B fragment: column-major W, same K order as A
        v8bf blo = *(const v8bf*)(bcol + k0 + ksel * 8);
        v8bf bhi = *(const v8bf*)(bcol + k0 + 16 + ksel * 8);
        v16bf b;
#pragma unroll
        for (int i = 0; i < 8; ++i) { b[i] = blo[i]; b[i + 8] = bhi[i]; }
        acc = __builtin_amdgcn_wmma_f32_16x16x32_bf16(false, a, false, b,
                                                      (short)0, acc, false, false);
    }
    // D layout: lanes 0-15 -> N=lane, M = r; lanes 16-31 -> N=lane-16, M = 8+r
    const int nn = lane & 15;
    const int mb = (lane >> 4) * 8;
#pragma unroll
    for (int r = 0; r < 8; ++r)
        outp[(size_t)(m0 + mb + r) * 128 + n0 + nn] = acc[r];
}

// ---------------- layer-2 small GEMM: [N x 128] @ [128 x 2] x3 ----------------
__global__ void gemm_small(const float* __restrict__ h,
                           const float* __restrict__ Wq, const float* __restrict__ Wk,
                           const float* __restrict__ Wv,
                           float* __restrict__ q, float* __restrict__ k,
                           float* __restrict__ v, int n) {
    __shared__ float sW[128 * 6];  // [k][j]: j=0,1 q ; 2,3 k ; 4,5 v
    for (int i = threadIdx.x; i < 128 * 2; i += blockDim.x) {
        int kk = i >> 1, j = i & 1;
        sW[kk * 6 + j]     = Wq[i];
        sW[kk * 6 + 2 + j] = Wk[i];
        sW[kk * 6 + 4 + j] = Wv[i];
    }
    __syncthreads();
    int node = blockIdx.x * blockDim.x + threadIdx.x;
    if (node >= n) return;
    const float4* hv = (const float4*)(h + (size_t)node * 128);
    float acc[6] = {0, 0, 0, 0, 0, 0};
#pragma unroll 4
    for (int c = 0; c < 32; ++c) {
        float4 x = hv[c];
        const float* w0 = &sW[(c * 4 + 0) * 6];
        const float* w1 = &sW[(c * 4 + 1) * 6];
        const float* w2 = &sW[(c * 4 + 2) * 6];
        const float* w3 = &sW[(c * 4 + 3) * 6];
#pragma unroll
        for (int j = 0; j < 6; ++j)
            acc[j] += x.x * w0[j] + x.y * w1[j] + x.z * w2[j] + x.w * w3[j];
    }
    q[node * 2 + 0] = acc[0]; q[node * 2 + 1] = acc[1];
    k[node * 2 + 0] = acc[2]; k[node * 2 + 1] = acc[3];
    v[node * 2 + 0] = acc[4]; v[node * 2 + 1] = acc[5];
}

// ---------------- edge attention: scores + exp + denom ----------------
template <int H, int C>
__global__ void edge_attn(const int* __restrict__ ei, const float* __restrict__ ea,
                          const float* __restrict__ q, const float* __restrict__ k,
                          const float* __restrict__ tbias, const float* __restrict__ freq,
                          const float* __restrict__ Ew, const float* __restrict__ Eb,
                          const unsigned* __restrict__ red,
                          float* __restrict__ ex, float* __restrict__ denom, int ne) {
    constexpr int HC = H * C;
    __shared__ float sEw[69 * H];
    __shared__ float sEb[H], sTb[H], sFreq[32];
    for (int i = threadIdx.x; i < 69 * H; i += blockDim.x) sEw[i] = Ew[i];
    if (threadIdx.x < H) { sEb[threadIdx.x] = Eb[threadIdx.x]; sTb[threadIdx.x] = tbias[threadIdx.x]; }
    if (threadIdx.x < 32) sFreq[threadIdx.x] = freq[threadIdx.x];
    __syncthreads();

    int e = blockIdx.x * blockDim.x + threadIdx.x;
    if (e >= ne) return;
    int src = ei[e], dst = ei[ne + e];

    const float rsC = rsqrtf((float)C);
    float attn[H];
#pragma unroll
    for (int h = 0; h < H; ++h) {
        float s = 0.0f;
        if constexpr (C % 4 == 0) {
            const float4* qp = (const float4*)(q + (size_t)src * HC + h * C);
            const float4* kp = (const float4*)(k + (size_t)dst * HC + h * C);
#pragma unroll
            for (int c = 0; c < C / 4; ++c) {
                float4 a = qp[c], b = kp[c];
                s += a.x * b.x + a.y * b.y + a.z * b.z + a.w * b.w;
            }
        } else {
#pragma unroll
            for (int c = 0; c < C; ++c)
                s += q[(size_t)src * HC + h * C + c] * k[(size_t)dst * HC + h * C + c];
        }
        attn[h] = s * rsC;
    }

    // time normalization
    float t = ea[(size_t)e * 5 + 1];
    float tmin = unorderf(red[0]), tmax = unorderf(red[1]);
    float tn = (tmax > tmin) ? (t - tmin) / (tmax - tmin + 1e-6f) : 1.0f;

    float acc[H];
#pragma unroll
    for (int h = 0; h < H; ++h) acc[h] = sEb[h] + sTb[h] * tn;

    // edge_attr part of projection
#pragma unroll
    for (int d = 0; d < 5; ++d) {
        float val = ea[(size_t)e * 5 + d];
#pragma unroll
        for (int h = 0; h < H; ++h) acc[h] += val * sEw[d * H + h];
    }
    // time-encoding part
    float la = log1pf(t + 1e-6f);
#pragma unroll 4
    for (int j = 0; j < 32; ++j) {
        float ang = la * sFreq[j];
        float sn = sinf(ang), cs = cosf(ang);
        const float* wrow = &sEw[(5 + 2 * j) * H];
#pragma unroll
        for (int h = 0; h < H; ++h) acc[h] += sn * wrow[h] + cs * wrow[H + h];
    }

#pragma unroll
    for (int h = 0; h < H; ++h) {
        float a = attn[h] + acc[h];
        a = (a >= 0.0f) ? a : 0.2f * a;   // leaky relu
        float eh = expf(a);
        ex[(size_t)e * H + h] = eh;
        atomicAdd(&denom[(size_t)dst * H + h], eh);
    }
}

// ---------------- edge aggregation: out[dst] += (ex/denom) * v[src] ----------------
template <int H, int C>
__global__ void edge_aggr(const int* __restrict__ ei, const float* __restrict__ v,
                          const float* __restrict__ ex, const float* __restrict__ denom,
                          float* __restrict__ out, int ne) {
    constexpr int HC = H * C;
    int idx = blockIdx.x * blockDim.x + threadIdx.x;
    if (idx >= ne * H) return;
    int e = idx / H, h = idx % H;
    int src = ei[e], dst = ei[ne + e];
    float a = ex[(size_t)e * H + h] / (denom[(size_t)dst * H + h] + 1e-8f);
    const float* vp = v + (size_t)src * HC + h * C;
    float* op = out + (size_t)dst * HC + h * C;
#pragma unroll
    for (int c = 0; c < C; ++c) atomicAdd(&op[c], a * vp[c]);
}

// ---------------- launch ----------------
extern "C" void kernel_launch(void* const* d_in, const int* in_sizes, int n_in,
                              void* d_out, int out_size, void* d_ws, size_t ws_size,
                              hipStream_t stream) {
    const float* x   = (const float*)d_in[0];
    const int*   ei  = (const int*)d_in[1];
    const float* ea  = (const float*)d_in[2];
    // per-layer params: Wq,Wk,Wv,freq,tbias,Ew,Eb at 3+7l..
    auto P = [&](int l, int j) { return (const float*)d_in[3 + l * 7 + j]; };

    char* ws = (char*)d_ws;
    const size_t A = (size_t)K_N * 128 * 4;  // 25,600,000 B (256-aligned)
    float* f0    = (float*)(ws + 0 * A);
    float* f1    = (float*)(ws + 1 * A);
    float* q     = (float*)(ws + 2 * A);
    float* k     = (float*)(ws + 3 * A);
    float* v     = (float*)(ws + 4 * A);
    float* ex    = (float*)(ws + 5 * A);                 // E*8*4 == A
    unsigned short* xbf = (unsigned short*)(ws + 6 * A); // N*128*2
    unsigned short* wbf = (unsigned short*)(ws + 6 * A + (size_t)K_N * 128 * 2);
    float* denom = (float*)((char*)wbf + 3 * 128 * 128 * 2);   // N*8*4
    unsigned* red = (unsigned*)((char*)denom + (size_t)K_N * 8 * 4);

    const int NB_NF = (K_N * 128 + 255) / 256;   // N*128 elementwise
    const int NB_E  = (K_E + 255) / 256;         // per-edge
    float* dout = (float*)d_out;

    // ts min/max (shared by all layers)
    init_red<<<1, 1, 0, stream>>>(red);
    ts_minmax<<<NB_E, 256, 0, stream>>>(ea, red, K_E);

    // ---- layers 0 and 1 (H=8, C=16) ----
    const float* fin = x;
    float* fout = f0;
    for (int l = 0; l < 2; ++l) {
        conv_x_bf16<<<NB_NF, 256, 0, stream>>>(fin, xbf, K_N * 128);
        conv_w_bf16<<<(3 * 128 * 128 + 255) / 256, 256, 0, stream>>>(P(l, 0), P(l, 1), P(l, 2), wbf);
        gemm_qkv_wmma<<<dim3(K_N / 16, 3), 256, 0, stream>>>(xbf, wbf, q, k, v);

        zero_f<<<(K_N * 8 + 255) / 256, 256, 0, stream>>>(denom, K_N * 8);
        edge_attn<8, 16><<<NB_E, 256, 0, stream>>>(ei, ea, q, k, P(l, 4), P(l, 3),
                                                   P(l, 5), P(l, 6), red, ex, denom, K_E);
        zero_f<<<NB_NF, 256, 0, stream>>>(fout, K_N * 128);
        edge_aggr<8, 16><<<(K_E * 8 + 255) / 256, 256, 0, stream>>>(ei, v, ex, denom, fout, K_E);
        elu_inplace<<<NB_NF, 256, 0, stream>>>(fout, K_N * 128);

        fin = fout;
        fout = f1;
    }

    // ---- layer 2 (H=1, C=2) ----
    gemm_small<<<(K_N + 127) / 128, 128, 0, stream>>>(f0 == fin ? f0 : f1, P(2, 0), P(2, 1), P(2, 2),
                                                      q, k, v, K_N);
    zero_f<<<(K_N + 255) / 256, 256, 0, stream>>>(denom, K_N);
    edge_attn<1, 2><<<NB_E, 256, 0, stream>>>(ei, ea, q, k, P(2, 4), P(2, 3),
                                              P(2, 5), P(2, 6), red, ex, denom, K_E);
    zero_f<<<(K_N * 2 + 255) / 256, 256, 0, stream>>>(dout, K_N * 2);
    edge_aggr<1, 2><<<NB_E, 256, 0, stream>>>(ei, v, ex, denom, dout, K_E);
}
